// GAT_76192719831391
// MI455X (gfx1250) — matrix operand order repaired and checked by reading, not compiled, not run on hardware
//
#include <hip/hip_runtime.h>

#define N_NODES 100000
#define N_EDGES 1600000
#define NTILES  (N_NODES / 16)   // 6250 exactly
#define NEG_SLOPE 0.2f
#define TPW 4                    // node tiles per wave in the WMMA kernel

typedef __attribute__((ext_vector_type(16))) _Float16 v16h;
typedef __attribute__((ext_vector_type(8)))  float    v8f;

// ---------- order-preserving float<->int mapping for atomicMax on floats ----
__device__ __forceinline__ int f2ord(float f) {
  int i = __float_as_int(f);
  return (i >= 0) ? i : (i ^ 0x7fffffff);
}
__device__ __forceinline__ float ord2f(int o) {
  int i = (o >= 0) ? o : (o ^ 0x7fffffff);
  return __int_as_float(i);
}

// ---------- small setup kernels ---------------------------------------------
__global__ void gat_init_nodes(int* __restrict__ segmax, float* __restrict__ denom) {
  int i = blockIdx.x * blockDim.x + threadIdx.x;
  if (i < N_NODES) { segmax[i] = (int)0x807fffff /* ord(-inf) */; denom[i] = 0.0f; }
}

__global__ void gat_zero(float* __restrict__ p, int n) {
  int i = blockIdx.x * blockDim.x + threadIdx.x;
  if (i < n) p[i] = 0.0f;
}

// Pack 3 weight matrices [64x64] f32 into WMMA B-fragment order (f16):
// layout [w][ct][kc][lane][16 halves]; contiguous k order used here (same as A), B 32x16 f16:
// lane L holds col N=L&15; VGPR v holds K=(L>>4)*16 + 2v (+ kc*32), K+1.
__global__ void gat_pack_w(const float* __restrict__ W0, const float* __restrict__ W1,
                           const float* __restrict__ W2, _Float16* __restrict__ wpack)
{
  int t = blockIdx.x * blockDim.x + threadIdx.x;   // 3*4*2*32 = 768 threads
  if (t >= 768) return;
  const int lane = t & 31;
  const int kc   = (t >> 5) & 1;
  const int ct   = (t >> 6) & 3;
  const int w    = t >> 8;
  const float* W = (w == 0) ? W0 : (w == 1) ? W1 : W2;
  const int n  = lane & 15;
  const int hi = lane >> 4;
  _Float16* o = wpack + (size_t)t * 16;
#pragma unroll
  for (int v = 0; v < 8; ++v) {
    const int k = kc * 32 + hi * 16 + 2 * v;
    o[2 * v]     = (_Float16)W[(size_t)k * 64 + ct * 16 + n];
    o[2 * v + 1] = (_Float16)W[(size_t)(k + 1) * 64 + ct * 16 + n];
  }
}

// ---------- node projection: 3 GEMMs [N,64]x[64,64] via WMMA ----------------
__device__ __forceinline__ void pack8(v16h& a, int base, float4 u, float4 v) {
  a[base + 0] = (_Float16)u.x; a[base + 1] = (_Float16)u.y;
  a[base + 2] = (_Float16)u.z; a[base + 3] = (_Float16)u.w;
  a[base + 4] = (_Float16)v.x; a[base + 5] = (_Float16)v.y;
  a[base + 6] = (_Float16)v.z; a[base + 7] = (_Float16)v.w;
}

// One wave handles TPW 16-row node tiles; B fragments loaded once per (w,ct)
// as packed v16h (2x global_load_b128) and reused for all tiles.
// A fragment per lane = two contiguous 8-float runs of the x row (b128 loads).
// C/D 16x16 f32: lane L holds col N=L&15; VGPR v: row M = v + 8*(L>>4).
// Tail handling: tile indices are CLAMPED (no divergent guards) so EXEC stays
// all-ones around every WMMA as the ISA requires; the one tail wave just
// recomputes/overwrites the last valid tile with identical values.
__global__ __launch_bounds__(256) void gat_proj(
    const float* __restrict__ X,
    const _Float16* __restrict__ wpack,   // [3][4][2][32][16] halves
    const float* __restrict__ blin,
    float* __restrict__ hs, float* __restrict__ hd, float* __restrict__ hl)
{
  const int lane  = threadIdx.x & 31;
  const int wave  = (int)((blockIdx.x * blockDim.x + threadIdx.x) >> 5);
  const int tile0 = wave * TPW;
  if (tile0 >= NTILES) return;            // whole-wave uniform exit

  int tiles[TPW];
#pragma unroll
  for (int t = 0; t < TPW; ++t) {
    const int tt = tile0 + t;
    tiles[t] = (tt < NTILES) ? tt : (NTILES - 1);   // clamp tail (wave-uniform)
  }

  const int mr = lane & 15;
  const int lo = lane >> 4;
  const int n  = lane & 15;

  v16h a0[TPW], a1[TPW];
#pragma unroll
  for (int t = 0; t < TPW; ++t) {
    const float* xrow = X + (size_t)(tiles[t] * 16 + mr) * 64;
    float4 f0 = *(const float4*)(xrow + lo * 16);
    float4 f1 = *(const float4*)(xrow + lo * 16 + 4);
    float4 f2 = *(const float4*)(xrow + lo * 16 + 8);
    float4 f3 = *(const float4*)(xrow + lo * 16 + 12);
    float4 f4 = *(const float4*)(xrow + 32 + lo * 16);
    float4 f5 = *(const float4*)(xrow + 32 + lo * 16 + 4);
    float4 f6 = *(const float4*)(xrow + 32 + lo * 16 + 8);
    float4 f7 = *(const float4*)(xrow + 32 + lo * 16 + 12);
    pack8(a0[t], 0, f0, f1);
    pack8(a0[t], 8, f2, f3);
    pack8(a1[t], 0, f4, f5);
    pack8(a1[t], 8, f6, f7);
  }

  float* Omat[3] = { hs, hd, hl };
  const v16h* wp = (const v16h*)wpack;

#pragma unroll
  for (int w = 0; w < 3; ++w) {
    float* O = Omat[w];
#pragma unroll
    for (int ct = 0; ct < 4; ++ct) {
      const int fbase = ((w * 4 + ct) * 2) * 32 + lane;
      v16h b0 = wp[fbase];        // kc = 0
      v16h b1 = wp[fbase + 32];   // kc = 1
      const float bias = (w == 2) ? blin[ct * 16 + n] : 0.0f;
#pragma unroll
      for (int t = 0; t < TPW; ++t) {
        v8f c = {};
        c = __builtin_amdgcn_wmma_f32_16x16x32_f16(false, a0[t], false, b0,
                                                   (short)0, c, false, false);
        c = __builtin_amdgcn_wmma_f32_16x16x32_f16(false, a1[t], false, b1,
                                                   (short)0, c, false, false);
#pragma unroll
        for (int v = 0; v < 8; ++v) {
          const float val = (w == 2) ? (c[v] + bias) : c[v];  // w is compile-time
          O[(size_t)(tiles[t] * 16 + lo * 8 + v) * 64 + ct * 16 + n] = val;
        }
      }
    }
  }
}

// ---------- alpha_{src,dst}[i] = <h_{src,dst}[i,:], a_{s,d}> ----------------
__global__ void gat_alpha(const float* __restrict__ hs, const float* __restrict__ hd,
                          const float* __restrict__ avs, const float* __restrict__ avd,
                          float* __restrict__ alps, float* __restrict__ alpd)
{
  int i = blockIdx.x * blockDim.x + threadIdx.x;
  if (i >= N_NODES) return;
  const float4* rs = (const float4*)(hs + (size_t)i * 64);
  const float4* rd = (const float4*)(hd + (size_t)i * 64);
  const float4* ws = (const float4*)avs;
  const float4* wd = (const float4*)avd;
  float s = 0.0f, d = 0.0f;
#pragma unroll
  for (int j = 0; j < 16; ++j) {
    float4 a = rs[j], w = ws[j];
    s += a.x * w.x + a.y * w.y + a.z * w.z + a.w * w.w;
    float4 b = rd[j], u = wd[j];
    d += b.x * u.x + b.y * u.y + b.z * u.z + b.w * u.w;
  }
  alps[i] = s;
  alpd[i] = d;
}

// ---------- edge phase ------------------------------------------------------
__device__ __forceinline__ float leaky(float v) { return (v > 0.0f) ? v : NEG_SLOPE * v; }

__global__ void gat_edge_max(const int* __restrict__ src, const int* __restrict__ dst,
                             const float* __restrict__ alps, const float* __restrict__ alpd,
                             int* __restrict__ segmax)
{
  int e = blockIdx.x * blockDim.x + threadIdx.x;
  if (e >= N_EDGES) return;
  int d = dst[e];
  float v = leaky(alps[src[e]] + alpd[d]);
  atomicMax(&segmax[d], f2ord(v));
}

__global__ void gat_edge_exp(const int* __restrict__ src, const int* __restrict__ dst,
                             const float* __restrict__ alps, const float* __restrict__ alpd,
                             const int* __restrict__ segmax,
                             float* __restrict__ z, float* __restrict__ denom)
{
  int e = blockIdx.x * blockDim.x + threadIdx.x;
  if (e >= N_EDGES) return;
  int d = dst[e];
  float v = leaky(alps[src[e]] + alpd[d]);
  float zz = __expf(v - ord2f(segmax[d]));
  z[e] = zz;
  atomicAdd(&denom[d], zz);
}

// one thread per (edge, channel): wave32 covers 32 consecutive floats of one
// destination row -> coalesced gathers + same-cacheline atomics (L2 resident).
__global__ void gat_edge_scatter(const int* __restrict__ src, const int* __restrict__ dst,
                                 const float* __restrict__ z, const float* __restrict__ denom,
                                 const float* __restrict__ hs, float* __restrict__ agg)
{
  long long t = (long long)blockIdx.x * blockDim.x + threadIdx.x;
  if (t >= (long long)N_EDGES * 64) return;
  int e = (int)(t >> 6);
  int c = (int)(t & 63);
  int s = src[e], d = dst[e];
  float coef = z[e] / (denom[d] + 1e-16f);
  atomicAdd(agg + ((size_t)d * 64 + c), coef * hs[(size_t)s * 64 + c]);
}

// ---------- combine: agg + bias + residual (+ReLU) --------------------------
__global__ void gat_combine(const float* __restrict__ agg, const float* __restrict__ bias,
                            const float* __restrict__ lin, float* __restrict__ out, int relu)
{
  int i = blockIdx.x * blockDim.x + threadIdx.x;
  if (i >= N_NODES * 64) return;
  float v = agg[i] + bias[i & 63] + lin[i];
  if (relu) v = fmaxf(v, 0.0f);
  out[i] = v;
}

// ---------- host orchestration ----------------------------------------------
extern "C" void kernel_launch(void* const* d_in, const int* in_sizes, int n_in,
                              void* d_out, int out_size, void* d_ws, size_t ws_size,
                              hipStream_t stream)
{
  const float* x   = (const float*)d_in[0];
  const int*   ei  = (const int*)d_in[1];
  const float* Ws1 = (const float*)d_in[2];
  const float* Wd1 = (const float*)d_in[3];
  const float* as1 = (const float*)d_in[4];
  const float* ad1 = (const float*)d_in[5];
  const float* b1  = (const float*)d_in[6];
  const float* Wl1 = (const float*)d_in[7];
  const float* bl1 = (const float*)d_in[8];
  const float* Ws2 = (const float*)d_in[9];
  const float* Wd2 = (const float*)d_in[10];
  const float* as2 = (const float*)d_in[11];
  const float* ad2 = (const float*)d_in[12];
  const float* b2  = (const float*)d_in[13];
  const float* Wl2 = (const float*)d_in[14];
  const float* bl2 = (const float*)d_in[15];
  const int* src = ei;
  const int* dst = ei + N_EDGES;

  char* p = (char*)d_ws;
  auto carve = [&](size_t bytes) -> char* {
    char* r = p;
    p += (bytes + 255) & ~(size_t)255;
    return r;
  };
  float* hs    = (float*)carve((size_t)N_NODES * 64 * sizeof(float));
  float* hd    = (float*)carve((size_t)N_NODES * 64 * sizeof(float));
  float* hl    = (float*)carve((size_t)N_NODES * 64 * sizeof(float));
  float* agg   = (float*)carve((size_t)N_NODES * 64 * sizeof(float));
  float* h1    = (float*)carve((size_t)N_NODES * 64 * sizeof(float));
  float* alps  = (float*)carve((size_t)N_NODES * sizeof(float));
  float* alpd  = (float*)carve((size_t)N_NODES * sizeof(float));
  int*   segmx = (int*)  carve((size_t)N_NODES * sizeof(int));
  float* denom = (float*)carve((size_t)N_NODES * sizeof(float));
  float* z     = (float*)carve((size_t)N_EDGES * sizeof(float));
  _Float16* wpack1 = (_Float16*)carve((size_t)768 * 16 * sizeof(_Float16)); // 24 KB
  _Float16* wpack2 = (_Float16*)carve((size_t)768 * 16 * sizeof(_Float16)); // 24 KB

  gat_pack_w<<<3, 256, 0, stream>>>(Ws1, Wd1, Wl1, wpack1);
  gat_pack_w<<<3, 256, 0, stream>>>(Ws2, Wd2, Wl2, wpack2);

  const int proj_waves  = (NTILES + TPW - 1) / TPW;          // 1563
  const int proj_blocks = (proj_waves + 7) / 8;              // 8 waves / 256-thr block

  auto run_layer = [&](const float* Xin, const _Float16* wpk,
                       const float* av_s, const float* av_d,
                       const float* bias, const float* blin,
                       float* outbuf, int relu) {
    gat_init_nodes<<<(N_NODES + 255) / 256, 256, 0, stream>>>(segmx, denom);
    gat_zero<<<(N_NODES * 64 + 255) / 256, 256, 0, stream>>>(agg, N_NODES * 64);
    gat_proj<<<proj_blocks, 256, 0, stream>>>(Xin, wpk, blin, hs, hd, hl);
    gat_alpha<<<(N_NODES + 255) / 256, 256, 0, stream>>>(hs, hd, av_s, av_d, alps, alpd);
    gat_edge_max<<<(N_EDGES + 255) / 256, 256, 0, stream>>>(src, dst, alps, alpd, segmx);
    gat_edge_exp<<<(N_EDGES + 255) / 256, 256, 0, stream>>>(src, dst, alps, alpd, segmx, z, denom);
    long long ntt = (long long)N_EDGES * 64;
    gat_edge_scatter<<<(unsigned)((ntt + 255) / 256), 256, 0, stream>>>(src, dst, z, denom, hs, agg);
    gat_combine<<<(N_NODES * 64 + 255) / 256, 256, 0, stream>>>(agg, bias, hl, outbuf, relu);
  };

  // layer 1: h1 = relu(GAT(x) + x@Wl1 + bl1)
  run_layer(x, wpack1, as1, ad1, b1, bl1, h1, 1);
  // layer 2: out = GAT(h1) + h1@Wl2 + bl2
  run_layer(h1, wpack2, as2, ad2, b2, bl2, (float*)d_out, 0);
}
